// TransformerAttention_12120397710003
// MI455X (gfx1250) — matrix operand order repaired and checked by reading, not compiled, not run on hardware
//
#include <hip/hip_runtime.h>
#include <hip/hip_bf16.h>
#include <cstdint>

typedef __attribute__((ext_vector_type(16))) _Float16 v16h;
typedef __attribute__((ext_vector_type(8)))  _Float16 v8h;
typedef __attribute__((ext_vector_type(4)))  _Float16 v4h;
typedef __attribute__((ext_vector_type(8)))  float    v8f;
typedef __attribute__((ext_vector_type(4)))  float    v4f;

#if __has_builtin(__builtin_amdgcn_tensor_load_to_lds) && \
    __has_builtin(__builtin_amdgcn_s_wait_tensorcnt)
#define HAVE_TDM 1
#else
#define HAVE_TDM 0
#endif

namespace {
constexpr int kB       = 8;
constexpr int kC       = 512;
constexpr int kN       = 1024;           // H*W
constexpr int kNH      = 8;
constexpr int kDH      = 64;
constexpr int kCols    = kB * kN;        // 8192 GEMM columns
constexpr int kQKVRows = 3 * kC;         // 1536
}

static __device__ __forceinline__ v8f zero8() {
  v8f z = {0.f, 0.f, 0.f, 0.f, 0.f, 0.f, 0.f, 0.f};
  return z;
}

static __device__ __forceinline__ v8f wmma_f16(v16h a, v16h b, v8f c) {
  // D = A(16x32 f16) * B(32x16 f16) + C(16x16 f32)
  return __builtin_amdgcn_wmma_f32_16x16x32_f16(false, a, false, b, (short)0, c,
                                                false, false);
}

// A fragment 16x32 f16 from LDS stored row-major [row][k].
// ISA: lane m=l&15; halves 0..7 -> K=kb..kb+7, halves 8..15 -> K=kb+16..kb+23,
// kb=(l>>4)*8.  Compiles to 2x ds_load_b128 when the row pitch is 16B aligned.
static __device__ __forceinline__ v16h frag_a(const _Float16* s, int stride,
                                              int row0, int lane, int k0) {
  const int m  = lane & 15;
  const int kb = (lane >> 4) * 8;
  const _Float16* rp = s + (row0 + m) * stride + k0 + kb;
  v16h a;
#pragma unroll
  for (int j = 0; j < 8; ++j) a[j] = rp[j];
#pragma unroll
  for (int j = 0; j < 8; ++j) a[8 + j] = rp[16 + j];
  return a;
}

// B fragment 32x16 f16 from LDS stored [col][k] (K contiguous per column).
// ISA: lane n=l&15; halves 0..7 -> K=kb..kb+7, halves 8..15 -> K=kb+16..kb+23,
// kb=(l>>4)*8.
static __device__ __forceinline__ v16h frag_b(const _Float16* s, int stride,
                                              int col0, int lane, int k0) {
  const int n  = lane & 15;
  const int kb = (lane >> 4) * 8;
  const _Float16* rp = s + (col0 + n) * stride + k0 + kb;
  v16h b;
#pragma unroll
  for (int j = 0; j < 8; ++j) b[j] = rp[j];
#pragma unroll
  for (int j = 0; j < 8; ++j) b[8 + j] = rp[16 + j];
  return b;
}

#if HAVE_TDM
typedef __attribute__((ext_vector_type(4))) unsigned int v4u;
typedef __attribute__((ext_vector_type(8))) int          v8i;
typedef __attribute__((ext_vector_type(4))) int          v4i;

// TDM: DMA a 64(rows) x 64(contiguous f16) tile from global into LDS,
// padding each 128B row by 4 dwords -> LDS row pitch 144B (= 72 halves).
static __device__ __forceinline__ void tdm_load_tile_64x64_f16(
    void* lds_dst, const void* gsrc, unsigned row_stride_elems) {
  unsigned lds          = (unsigned)(size_t)lds_dst;  // low 32b = LDS offset
  unsigned long long ga = (unsigned long long)(size_t)gsrc;
  v4u g0 = {1u,  // count=1 (valid), is_restore=0, gather_mode=0
            lds,
            (unsigned)(ga & 0xffffffffu),
            (unsigned)(ga >> 32) | (2u << 30)};  // type=2 ("image")
  // data_size=1 (2B), pad_enable=1, pad_interval=4 (128B), pad_amount=3 (4dw)
  v8i g1 = {(int)0x07110000,
            (int)(64u << 16),       // tensor_dim0 = 64
            (int)(64u << 16),       // tensor_dim1 = 64
            (int)(64u << 16),       // tile_dim0  = 64
            64,                     // tile_dim1  = 64 (tile_dim2 = 0)
            (int)row_stride_elems,  // tensor_dim0_stride (data_size units)
            0, 0};
  v4i gz = {0, 0, 0, 0};
#if __clang_major__ >= 23
  v8i gz8 = {0, 0, 0, 0, 0, 0, 0, 0};
  __builtin_amdgcn_tensor_load_to_lds(g0, g1, gz, gz, gz8, 0);
#else
  __builtin_amdgcn_tensor_load_to_lds(g0, g1, gz, gz, 0);
#endif
}
#endif

// ---------------------------------------------------------------------------
// Kernel 1: qkv = W_qkv[1536,512] @ x[b][512,1024]  ->  f16 qkvh[1536, 8192]
// Software-pipelined: next tile's global loads issued before the WMMAs,
// registers drained into the alternate LDS buffer after the compute.
// ---------------------------------------------------------------------------
__global__ __launch_bounds__(256) void qkv_gemm_kernel(
    const float* __restrict__ X, const float* __restrict__ Wqkv,
    _Float16* __restrict__ qkvh) {
  constexpr int LDT = 40;  // 32 K + pad, 80B rows (16B aligned, bank-spread)
  __shared__ _Float16 As[2][128 * LDT];
  __shared__ _Float16 Bs[2][128 * LDT];
  const int tid  = threadIdx.x;
  const int lane = tid & 31;
  const int wave = tid >> 5;
  const int o0 = blockIdx.x * 128;
  const int b  = blockIdx.y >> 3;
  const int n0 = (blockIdx.y & 7) * 128;
  const size_t xbase = (size_t)b * (kC * kN);

  v8f acc[8];
#pragma unroll
  for (int i = 0; i < 8; ++i) acc[i] = zero8();

  v4f areg[4], breg[4];
  auto load_tile = [&](int k0) {
#pragma unroll
    for (int i = 0; i < 4; ++i) {
      int e = tid + i * 256;
      int r = e >> 3, kc = (e & 7) * 4;
      areg[i] = *(const v4f*)(Wqkv + (size_t)(o0 + r) * kC + k0 + kc);
    }
#pragma unroll
    for (int i = 0; i < 4; ++i) {
      int e = tid + i * 256;
      int nnc = (e & 31) * 4, kk = e >> 5;
      breg[i] = *(const v4f*)(X + xbase + (size_t)(k0 + kk) * kN + n0 + nnc);
    }
  };
  auto store_tile = [&](int buf) {
#pragma unroll
    for (int i = 0; i < 4; ++i) {
      int e = tid + i * 256;
      int r = e >> 3, kc = (e & 7) * 4;
      v4h h4;
#pragma unroll
      for (int j = 0; j < 4; ++j) h4[j] = (_Float16)areg[i][j];
      *(v4h*)(&As[buf][r * LDT + kc]) = h4;
    }
#pragma unroll
    for (int i = 0; i < 4; ++i) {
      int e = tid + i * 256;
      int nnc = (e & 31) * 4, kk = e >> 5;
#pragma unroll
      for (int j = 0; j < 4; ++j)
        Bs[buf][(nnc + j) * LDT + kk] = (_Float16)breg[i][j];
    }
  };

  load_tile(0);
  store_tile(0);
  for (int kt = 0; kt < 16; ++kt) {
    const int cur = kt & 1;
    __syncthreads();  // publish As/Bs[cur]; frees buffer cur^1 for writing
    if (kt + 1 < 16) load_tile((kt + 1) * 32);  // in flight through the WMMAs
    v16h a = frag_a(As[cur], LDT, wave * 16, lane, 0);
#pragma unroll
    for (int nt = 0; nt < 8; ++nt) {
      v16h bb = frag_b(Bs[cur], LDT, nt * 16, lane, 0);
      acc[nt] = wmma_f16(a, bb, acc[nt]);
    }
    if (kt + 1 < 16) store_tile(cur ^ 1);
  }

  const int hi = lane >> 4, nl = lane & 15;
#pragma unroll
  for (int nt = 0; nt < 8; ++nt)
#pragma unroll
    for (int r = 0; r < 8; ++r) {
      int o   = o0 + wave * 16 + r + hi * 8;  // C-frag: M = r (+8 for hi half)
      int col = b * kN + n0 + nt * 16 + nl;
      qkvh[(size_t)o * kCols + col] = (_Float16)acc[nt][r];
    }
}

// ---------------------------------------------------------------------------
// Kernel 2: flash attention per (b, h, 128-query block); writes f16 o in the
// torch raw-reshape order: obufh[b*N*C + i*C + h*64 + d].
// K double-buffered via registers; V double-buffered via TDM (the two V
// buffers alias the Q staging area, which is dead after the Q A-fragments
// are hoisted into registers).
// ---------------------------------------------------------------------------
__global__ __launch_bounds__(256) void attn_kernel(
    const _Float16* __restrict__ qkvh, _Float16* __restrict__ obufh) {
  constexpr int LQ = 72;  // 64 + pad, 144B rows (16B aligned, conflict-free)
  __shared__ _Float16 Qbuf[128 * LQ];   // Q staging, then 2x V tiles
  __shared__ _Float16 Ks[2][64 * LQ];   // [key j][d]
  __shared__ _Float16 Ps[8 * 16 * LQ];  // per-wave P round-trip
  const int tid = threadIdx.x, lane = tid & 31, wave = tid >> 5;
  const int b = blockIdx.x, h = blockIdx.y, i0 = blockIdx.z * 128;
  const int hi = lane >> 4, nl = lane & 15;

  // V tiles alias the Q staging buffer: Vt(buf) = Qbuf + buf*64*LQ
  // (no pointer array: constant-addrspacecast array initializers don't link)
  const _Float16* kbase = qkvh + (size_t)(kC + h * kDH) * kCols + b * kN;
  const _Float16* vbase = qkvh + (size_t)(2 * kC + h * kDH) * kCols + b * kN;

  // Stage Q [i][d]: b128 global loads over i, b16 transpose scatter.
  for (int e = tid; e < 16 * 64; e += 256) {
    int iic = (e & 15) * 8, d = e >> 4;
    v8h q = *(const v8h*)(qkvh + (size_t)(h * kDH + d) * kCols + b * kN + i0 +
                          iic);
#pragma unroll
    for (int j = 0; j < 8; ++j) Qbuf[(iic + j) * LQ + d] = q[j];
  }
  __syncthreads();
  // Q A-fragments are loop-invariant: hoist into registers.
  v16h qa0 = frag_a(Qbuf, LQ, wave * 16, lane, 0);
  v16h qa1 = frag_a(Qbuf, LQ, wave * 16, lane, 32);
  __syncthreads();  // all waves done with Q before V DMA reuses the space

  v8h kreg[2];
  auto kload = [&](int j0) {
#pragma unroll
    for (int i = 0; i < 2; ++i) {
      int e = tid + i * 256;
      int jjc = (e & 7) * 8, d = e >> 3;
      kreg[i] = *(const v8h*)(kbase + (size_t)d * kCols + j0 + jjc);
    }
  };
  auto kstore = [&](int buf) {
#pragma unroll
    for (int i = 0; i < 2; ++i) {
      int e = tid + i * 256;
      int jjc = (e & 7) * 8, d = e >> 3;
#pragma unroll
      for (int j = 0; j < 8; ++j) Ks[buf][(jjc + j) * LQ + d] = kreg[i][j];
    }
  };
#if !HAVE_TDM
  auto vstage = [&](_Float16* dst, int j0) {
    for (int e = tid; e < 8 * 64; e += 256) {
      int jjc = (e & 7) * 8, d = e >> 3;
      *(v8h*)(dst + d * LQ + jjc) =
          *(const v8h*)(vbase + (size_t)d * kCols + j0 + jjc);
    }
  };
#endif

  // Prologue: block 0 staged into buffer 0 (Vt(0) == Qbuf).
#if HAVE_TDM
  if (wave == 0) tdm_load_tile_64x64_f16(Qbuf, vbase, kCols);
#endif
  kload(0);
  kstore(0);
#if !HAVE_TDM
  vstage(Qbuf, 0);
#endif
#if HAVE_TDM
  if (wave == 0) __builtin_amdgcn_s_wait_tensorcnt(0);
#endif

  v8f oacc[4];
#pragma unroll
  for (int i = 0; i < 4; ++i) oacc[i] = zero8();
  float mrow[8], lrow[8];
#pragma unroll
  for (int r = 0; r < 8; ++r) { mrow[r] = -1e30f; lrow[r] = 0.f; }
  _Float16* Pw = Ps + wave * 16 * LQ;

  for (int blk = 0; blk < 16; ++blk) {
    const int cur = blk & 1;
    _Float16* Vcur = Qbuf + (cur ? 64 * LQ : 0);
    _Float16* Vnxt = Qbuf + (cur ? 0 : 64 * LQ);
    __syncthreads();  // publish Ks[cur], Vcur; frees the other buffers
    if (blk + 1 < 16) {
#if HAVE_TDM
      if (wave == 0)
        tdm_load_tile_64x64_f16(Vnxt, vbase + (blk + 1) * 64, kCols);
#endif
      kload((blk + 1) * 64);  // global loads in flight through the compute
    }

    // S = (Q K^T) * dh^-0.5  for this wave's 16 query rows x 64 keys
    v8f sacc[4];
#pragma unroll
    for (int ct = 0; ct < 4; ++ct) {
      v8f s = zero8();
      s = wmma_f16(qa0, frag_b(Ks[cur], LQ, ct * 16, lane, 0), s);
      s = wmma_f16(qa1, frag_b(Ks[cur], LQ, ct * 16, lane, 32), s);
#pragma unroll
      for (int r = 0; r < 8; ++r) s[r] *= 0.125f;  // 1/sqrt(64)
      sacc[ct] = s;
    }

    // Online softmax; C layout: VGPR r holds row r (lanes 0-15) / r+8 (16-31).
    float mnew[8];
#pragma unroll
    for (int r = 0; r < 8; ++r) {
      float v = fmaxf(fmaxf(sacc[0][r], sacc[1][r]),
                      fmaxf(sacc[2][r], sacc[3][r]));
#pragma unroll
      for (int msk = 1; msk <= 8; msk <<= 1)
        v = fmaxf(v, __shfl_xor(v, msk, 32));
      mnew[r] = fmaxf(mrow[r], v);
    }
    float alpha[8], rsum[8];
#pragma unroll
    for (int r = 0; r < 8; ++r) {
      alpha[r] = __expf(mrow[r] - mnew[r]);
      rsum[r]  = 0.f;
    }
#pragma unroll
    for (int ct = 0; ct < 4; ++ct)
#pragma unroll
      for (int r = 0; r < 8; ++r) {
        float p = __expf(sacc[ct][r] - mnew[r]);
        Pw[(r + 8 * hi) * LQ + ct * 16 + nl] = (_Float16)p;  // C->A via LDS
        rsum[r] += p;
      }
#pragma unroll
    for (int r = 0; r < 8; ++r) {
      float v = rsum[r];
#pragma unroll
      for (int msk = 1; msk <= 8; msk <<= 1) v += __shfl_xor(v, msk, 32);
      lrow[r] = lrow[r] * alpha[r] + v;
      mrow[r] = mnew[r];
    }
#pragma unroll
    for (int dt = 0; dt < 4; ++dt)
#pragma unroll
      for (int r = 0; r < 8; ++r) oacc[dt][r] *= alpha[r];

    // O += P @ V  (per-wave LDS ops are in-order, no cross-wave hazard on Pw)
#pragma unroll
    for (int dt = 0; dt < 4; ++dt)
#pragma unroll
      for (int kt = 0; kt < 2; ++kt) {
        v16h a  = frag_a(Pw, LQ, 0, lane, kt * 32);
        v16h bb = frag_b(Vcur, LQ, dt * 16, lane, kt * 32);
        oacc[dt] = wmma_f16(a, bb, oacc[dt]);
      }

    if (blk + 1 < 16) {
      kstore(cur ^ 1);
#if !HAVE_TDM
      vstage(Vnxt, (blk + 1) * 64);
#endif
#if HAVE_TDM
      if (wave == 0) __builtin_amdgcn_s_wait_tensorcnt(0);
#endif
    }
  }

#pragma unroll
  for (int dt = 0; dt < 4; ++dt)
#pragma unroll
    for (int r = 0; r < 8; ++r) {
      float v = oacc[dt][r] / lrow[r];
      int ig  = i0 + wave * 16 + r + 8 * hi;
      int d   = dt * 16 + nl;
      obufh[(size_t)b * (kN * kC) + (size_t)ig * kC + h * kDH + d] =
          (_Float16)v;
    }
}

// ---------------------------------------------------------------------------
// Kernel 3: out0 = W_proj @ o_reshaped + b_proj  (f32 into d_out)
// ---------------------------------------------------------------------------
__global__ __launch_bounds__(256) void proj_gemm_kernel(
    const _Float16* __restrict__ obufh, const float* __restrict__ Wproj,
    const float* __restrict__ bproj, float* __restrict__ out) {
  constexpr int LDT = 40;
  __shared__ _Float16 As[2][128 * LDT];
  __shared__ _Float16 Bs[2][128 * LDT];
  const int tid = threadIdx.x, lane = tid & 31, wave = tid >> 5;
  const int o0 = blockIdx.x * 128;
  const int b  = blockIdx.y >> 3;
  const int n0 = (blockIdx.y & 7) * 128;
  const size_t obase = (size_t)b * (kC * kN);

  v8f acc[8];
#pragma unroll
  for (int i = 0; i < 8; ++i) acc[i] = zero8();

  v4f areg[4];
  v8h breg[2];
  auto load_tile = [&](int k0) {
#pragma unroll
    for (int i = 0; i < 4; ++i) {
      int e = tid + i * 256;
      int r = e >> 3, kc = (e & 7) * 4;
      areg[i] = *(const v4f*)(Wproj + (size_t)(o0 + r) * kC + k0 + kc);
    }
#pragma unroll
    for (int i = 0; i < 2; ++i) {
      int e = tid + i * 256;
      int nnc = (e & 15) * 8, kk = e >> 4;
      breg[i] = *(const v8h*)(obufh + obase + (size_t)(k0 + kk) * kN + n0 + nnc);
    }
  };
  auto store_tile = [&](int buf) {
#pragma unroll
    for (int i = 0; i < 4; ++i) {
      int e = tid + i * 256;
      int r = e >> 3, kc = (e & 7) * 4;
      v4h h4;
#pragma unroll
      for (int j = 0; j < 4; ++j) h4[j] = (_Float16)areg[i][j];
      *(v4h*)(&As[buf][r * LDT + kc]) = h4;
    }
#pragma unroll
    for (int i = 0; i < 2; ++i) {
      int e = tid + i * 256;
      int nnc = (e & 15) * 8, kk = e >> 4;
#pragma unroll
      for (int j = 0; j < 8; ++j)
        Bs[buf][(nnc + j) * LDT + kk] = breg[i][j];
    }
  };

  load_tile(0);
  store_tile(0);
  for (int kt = 0; kt < 16; ++kt) {
    const int cur = kt & 1;
    __syncthreads();
    if (kt + 1 < 16) load_tile((kt + 1) * 32);
    v16h a = frag_a(As[cur], LDT, wave * 16, lane, 0);
#pragma unroll
    for (int nt = 0; nt < 8; ++nt) {
      v16h bb = frag_b(Bs[cur], LDT, nt * 16, lane, 0);
      acc[nt] = wmma_f16(a, bb, acc[nt]);
    }
    if (kt + 1 < 16) store_tile(cur ^ 1);
  }

  const int hi = lane >> 4, nl = lane & 15;
#pragma unroll
  for (int r = 0; r < 8; ++r) {
    int o      = o0 + wave * 16 + r + hi * 8;
    float bias = bproj[o];
#pragma unroll
    for (int nt = 0; nt < 8; ++nt) {
      int p = n0 + nt * 16 + nl;
      out[(size_t)b * (kC * kN) + (size_t)o * kN + p] = acc[nt][r] + bias;
    }
  }
}

// ---------------------------------------------------------------------------
// Kernel 4: fused spatial attention. h1 = relu(W_att1 @ out0 + b1) via WMMA,
// att = sigmoid(W_att2 @ h1 + b2); scale out0 in place; write spatial_att.
// Each block owns a disjoint (b, 128-pixel) slab for read & write.
// ---------------------------------------------------------------------------
__global__ __launch_bounds__(256) void spatial_att_kernel(
    const float* __restrict__ Watt1, const float* __restrict__ batt1,
    const float* __restrict__ Watt2, const float* __restrict__ batt2,
    float* __restrict__ out) {
  constexpr int LDT = 40;
  __shared__ _Float16 As[2][128 * LDT];
  __shared__ _Float16 Bs[2][128 * LDT];
  __shared__ float attb[128];
  const int tid = threadIdx.x, lane = tid & 31, wave = tid >> 5;
  const int b  = blockIdx.x;
  const int p0 = blockIdx.y * 128;
  const int hi = lane >> 4, nl = lane & 15;
  const size_t obase = (size_t)b * (kC * kN);

  if (tid < 128) attb[tid] = batt2[0];

  v8f acc[8];
#pragma unroll
  for (int i = 0; i < 8; ++i) acc[i] = zero8();

  v4f areg[4], breg[4];
  auto load_tile = [&](int k0) {
#pragma unroll
    for (int i = 0; i < 4; ++i) {
      int e = tid + i * 256;
      int r = e >> 3, kc = (e & 7) * 4;
      areg[i] = *(const v4f*)(Watt1 + (size_t)r * kC + k0 + kc);
    }
#pragma unroll
    for (int i = 0; i < 4; ++i) {
      int e = tid + i * 256;
      int nnc = (e & 31) * 4, kk = e >> 5;
      breg[i] = *(const v4f*)(out + obase + (size_t)(k0 + kk) * kN + p0 + nnc);
    }
  };
  auto store_tile = [&](int buf) {
#pragma unroll
    for (int i = 0; i < 4; ++i) {
      int e = tid + i * 256;
      int r = e >> 3, kc = (e & 7) * 4;
      v4h h4;
#pragma unroll
      for (int j = 0; j < 4; ++j) h4[j] = (_Float16)areg[i][j];
      *(v4h*)(&As[buf][r * LDT + kc]) = h4;
    }
#pragma unroll
    for (int i = 0; i < 4; ++i) {
      int e = tid + i * 256;
      int nnc = (e & 31) * 4, kk = e >> 5;
#pragma unroll
      for (int j = 0; j < 4; ++j)
        Bs[buf][(nnc + j) * LDT + kk] = (_Float16)breg[i][j];
    }
  };

  load_tile(0);
  store_tile(0);
  for (int kt = 0; kt < 16; ++kt) {
    const int cur = kt & 1;
    __syncthreads();
    if (kt + 1 < 16) load_tile((kt + 1) * 32);
    v16h a = frag_a(As[cur], LDT, wave * 16, lane, 0);
#pragma unroll
    for (int ct = 0; ct < 8; ++ct) {
      v16h bb = frag_b(Bs[cur], LDT, ct * 16, lane, 0);
      acc[ct] = wmma_f16(a, bb, acc[ct]);
    }
    if (kt + 1 < 16) store_tile(cur ^ 1);
  }

  // att2: sum_j w2[j] * relu(h1[j,p] + b1[j]); halves hold different rows.
#pragma unroll
  for (int ct = 0; ct < 8; ++ct) {
    float part = 0.f;
#pragma unroll
    for (int r = 0; r < 8; ++r) {
      int j    = wave * 16 + r + 8 * hi;
      float hv = acc[ct][r] + batt1[j];
      hv       = fmaxf(hv, 0.f);
      part += Watt2[j] * hv;
    }
    part += __shfl_xor(part, 16, 32);  // combine the two row-halves (same p)
    if (lane < 16) atomicAdd(&attb[ct * 16 + nl], part);  // ds_add_f32
  }
  __syncthreads();

  if (tid < 128) {
    float a = attb[tid];
    a         = 1.f / (1.f + __expf(-a));
    attb[tid] = a;
    // spatial_att output: second tuple member, appended after out (4194304 f32)
    out[(size_t)kB * kC * kN + (size_t)b * kN + p0 + tid] = a;
  }
  __syncthreads();

  for (int e = tid; e < kC * 128; e += 256) {
    int c = e >> 7, pp = e & 127;
    size_t idx = (size_t)b * (kC * kN) + (size_t)c * kN + p0 + pp;
    out[idx] *= attb[pp];
  }
}

// ---------------------------------------------------------------------------
extern "C" void kernel_launch(void* const* d_in, const int* in_sizes, int n_in,
                              void* d_out, int out_size, void* d_ws,
                              size_t ws_size, hipStream_t stream) {
  (void)in_sizes; (void)n_in; (void)out_size; (void)ws_size;
  const float* X     = (const float*)d_in[0];
  const float* Wqkv  = (const float*)d_in[1];
  const float* Wproj = (const float*)d_in[2];
  const float* bproj = (const float*)d_in[3];
  const float* Watt1 = (const float*)d_in[4];
  const float* batt1 = (const float*)d_in[5];
  const float* Watt2 = (const float*)d_in[6];
  const float* batt2 = (const float*)d_in[7];
  float* out = (float*)d_out;

  // Workspace: f16 qkv [1536, 8192] (24 MiB) + f16 o [B*N, C] (8 MiB)
  _Float16* qkvh  = (_Float16*)d_ws;
  _Float16* obufh = (_Float16*)((char*)d_ws +
                                (size_t)kQKVRows * kCols * sizeof(_Float16));

  qkv_gemm_kernel<<<dim3(kQKVRows / 128, kCols / 128), 256, 0, stream>>>(
      X, Wqkv, qkvh);
  attn_kernel<<<dim3(kB, kNH, kN / 128), 256, 0, stream>>>(qkvh, obufh);
  proj_gemm_kernel<<<dim3(kC / 128, kCols / 128), 256, 0, stream>>>(
      obufh, Wproj, bproj, out);
  spatial_att_kernel<<<dim3(kB, kN / 128), 256, 0, stream>>>(
      Watt1, batt1, Watt2, batt2, out);
}
